// Decoder_60163901882966
// MI455X (gfx1250) — compile-verified, matrix operand order checked by reading, not run on hardware
//
#include <hip/hip_runtime.h>

typedef __attribute__((ext_vector_type(16))) __bf16 v16bf;
typedef __attribute__((ext_vector_type(8)))  __bf16 v8bf;
typedef __attribute__((ext_vector_type(8)))  float  v8f;
typedef __attribute__((ext_vector_type(4)))  unsigned v4u;
typedef __attribute__((ext_vector_type(8)))  int      v8i_;
typedef __attribute__((ext_vector_type(4)))  int      v4i_;

#define BSZ   64
#define TIN_  256
#define TOUT_ 400
#define NM_   80
#define BF_   512
#define MID_  128
#define HH    1024
#define PRE_  256
#define XS_STRIDE 136   // 128 + 8 pad (16B) to spread row readers across LDS banks

__device__ __forceinline__ float sigf(float x){ return 1.0f/(1.0f + __expf(-x)); }

// Build a 16x32 (or 32x16) bf16 WMMA fragment for wave32.
// Lane (l15, hi): elems 0..7 -> K = hi*8 + j ; elems 8..15 -> K = 16 + hi*8 + j.
__device__ __forceinline__ v16bf frag16(const __bf16* p, int hi){
  const __bf16* q = p + hi*8;
  v8bf lo = *(const v8bf*)(q);
  v8bf hv = *(const v8bf*)(q + 16);
  v16bf r;
#pragma unroll
  for (int i=0;i<8;++i){ r[i]=lo[i]; r[i+8]=hv[i]; }
  return r;
}

__device__ __forceinline__ v8f wmma_bf(v16bf a, v16bf b, v8f c){
  return __builtin_amdgcn_wmma_f32_16x16x32_bf16(false, a, false, b, (short)0, c, false, false);
}

__device__ __forceinline__ void wait_async0(){
  asm volatile("s_wait_asynccnt 0x0" ::: "memory");
}

// Async-copy one 64-byte strip per thread of A[64 x wA] chunk kc..kc+127 into an
// LDS panel, via GLOBAL_LOAD_ASYNC_TO_LDS_B128 (ASYNCcnt path, no VGPR staging).
__device__ __forceinline__ void stage_async(const __bf16* __restrict__ A, int wA, int kc,
                                            __bf16* dstbuf, int tid){
  int row = tid >> 2;
  int cb  = (tid & 3) * 32;
  const __bf16* src = A + (size_t)row*wA + kc + cb;
  __bf16* dst = dstbuf + row*XS_STRIDE + cb;
  unsigned dlo = (unsigned)(uintptr_t)(void*)dst;   // low 32 bits of LDS-aperture addr == LDS offset
  asm volatile("global_load_async_to_lds_b128 %0, %1, off"           :: "v"(dlo), "v"(src) : "memory");
  asm volatile("global_load_async_to_lds_b128 %0, %1, off offset:16" :: "v"(dlo), "v"(src) : "memory");
  asm volatile("global_load_async_to_lds_b128 %0, %1, off offset:32" :: "v"(dlo), "v"(src) : "memory");
  asm volatile("global_load_async_to_lds_b128 %0, %1, off offset:48" :: "v"(dlo), "v"(src) : "memory");
}

// TDM: one descriptor moves a 64-row x 128-col bf16 tile (row-padded +16B in LDS)
// from global to LDS. Tracked by TENSORcnt; issued by one wave only.
__device__ __forceinline__ void tdm_stage(const __bf16* __restrict__ A, int wA, int kc,
                                          __bf16* dstbuf){
  unsigned long long ga = (unsigned long long)(uintptr_t)(A + kc);
  unsigned ldsoff = (unsigned)(uintptr_t)(void*)dstbuf;
  v4u g0;
  g0[0] = 1u;                                            // count=1, user load descriptor
  g0[1] = ldsoff;                                        // lds_addr (bytes)
  g0[2] = (unsigned)(ga & 0xFFFFFFFFu);                  // global_addr[31:0]
  g0[3] = (unsigned)((ga >> 32) & 0x01FFFFFFu) | 0x80000000u; // global_addr[56:32] | type=2
  v8i_ g1;
  g1[0] = (int)((1u<<16) | (1u<<20) | (5u<<22) | (3u<<25)); // data_size=2B, pad_en, 64dw interval, 4dw pad
  g1[1] = (int)(((unsigned)wA & 0xFFFFu) << 16);         // tensor_dim0[15:0] @ bits 63:48
  g1[2] = (int)((((unsigned)wA >> 16) & 0xFFFFu) | (64u << 16)); // tensor_dim0[31:16] | tensor_dim1[15:0]=64
  g1[3] = (int)(128u << 16);                             // tensor_dim1[31:16]=0 | tile_dim0=128
  g1[4] = (int)64u;                                      // tile_dim1=64, tile_dim2=0
  g1[5] = (int)wA;                                       // tensor_dim0_stride[31:0]
  g1[6] = 0;                                             // stride[47:32] | tensor_dim1_stride[15:0]
  g1[7] = 0;
  v4i_ z4 = {0,0,0,0};
#if __clang_major__ >= 23
  v8i_ z8 = {0,0,0,0,0,0,0,0};
  __builtin_amdgcn_tensor_load_to_lds(g0, g1, z4, z4, z8, 0);
#else
  __builtin_amdgcn_tensor_load_to_lds(g0, g1, z4, z4, 0);
#endif
}

// One A-region GEMM with double-buffered LDS staging:
// chunk 0 via async-to-LDS (all waves, lowest latency), chunks 1..n-1 via TDM
// descriptors issued one iteration ahead by wave 0 (TENSORcnt pipeline).
// Must be called uniformly by all 256 threads. wA must be a multiple of 128.
__device__ __forceinline__ void gemm_region(
    const __bf16* __restrict__ A, int wA,
    const __bf16* __restrict__ W0, const __bf16* __restrict__ W1,
    __bf16* Xs0, __bf16* Xs1, int tid, int wv, int hi, int arow, v8f& acc0, v8f& acc1)
{
  int n = wA >> 7;
  stage_async(A, wA, 0, Xs0, tid);
  for (int i = 0; i < n; ++i){
    if (i + 1 < n && wv == 0) tdm_stage(A, wA, (i+1) << 7, (i & 1) ? Xs0 : Xs1);
    if (i == 0){
      wait_async0();
    } else if (i + 1 < n){
      __builtin_amdgcn_s_wait_tensorcnt((short)1);   // prev chunk landed, next still in flight
    } else {
      __builtin_amdgcn_s_wait_tensorcnt((short)0);
    }
    __syncthreads();
    int kc = i << 7;
    __builtin_prefetch(W0 + kc + 128, 0, 3);
    __builtin_prefetch(W1 + kc + 128, 0, 3);
    const __bf16* xrow = ((i & 1) ? Xs1 : Xs0) + arow*XS_STRIDE;
#pragma unroll
    for (int k = 0; k < 128; k += 32){
      v16bf a  = frag16(xrow + k, hi);
      v16bf b0 = frag16(W0 + kc + k, hi);
      v16bf b1 = frag16(W1 + kc + k, hi);
      acc0 = wmma_bf(a, b0, acc0);
      acc1 = wmma_bf(a, b1, acc1);
    }
    __syncthreads();
  }
}

__global__ void zero_kernel(float* p, int n){
  int i = blockIdx.x*blockDim.x + threadIdx.x;
  if (i < n) p[i] = 0.0f;
}

__global__ void cvt_kernel(const float* __restrict__ s, __bf16* __restrict__ d, int n){
  int i = blockIdx.x*blockDim.x + threadIdx.x;
  if (i < n) d[i] = (__bf16)s[i];
}

// PreNet for all timesteps: tgt[:, :, t] = (t==0) ? 0 : target[:, :, t-1]
__global__ void prenet_kernel(const float* __restrict__ target,
                              const float* __restrict__ W1, const float* __restrict__ b1,
                              const float* __restrict__ W2, const float* __restrict__ b2,
                              __bf16* __restrict__ xpre){
  int t = blockIdx.x >> 6;
  int b = blockIdx.x & 63;
  int p = threadIdx.x;
  __shared__ float x1s[256];
  float s = b1[p];
  if (t > 0){
    const float* tg = target + (size_t)b*NM_*TOUT_ + (t-1);
#pragma unroll 4
    for (int c=0;c<NM_;++c) s += W1[p*NM_ + c]*tg[c*TOUT_];
  }
  x1s[p] = fmaxf(s, 0.0f);
  __syncthreads();
  float s2 = b2[p];
  const float* w2 = W2 + p*256;
#pragma unroll 8
  for (int c=0;c<256;++c) s2 += w2[c]*x1s[c];
  xpre[((size_t)t*BSZ + b)*PRE_ + p] = (__bf16)fmaxf(s2, 0.0f);
}

// key_proj[b,ti,m] = input[b,ti,:] . Wk[m,:]
__global__ void key_kernel(const float* __restrict__ input, const float* __restrict__ Wk,
                           float* __restrict__ key){
  int b  = blockIdx.x >> 8;
  int ti = blockIdx.x & 255;
  int m  = threadIdx.x;
  const float* in = input + ((size_t)b*TIN_ + ti)*BF_;
  const float* w  = Wk + m*BF_;
  float s = 0.0f;
#pragma unroll 8
  for (int f=0; f<BF_; ++f) s += in[f]*w[f];
  key[((size_t)b*TIN_ + ti)*MID_ + m] = s;
}

// Fused LSTM step. grid = HH/16 = 64, block = 256.
// Wave w: M-tile = w&3, gate pair = w>>2  (two f32 accumulators per wave).
__global__ __launch_bounds__(256)
void lstm_kernel(const __bf16* __restrict__ A0, int w0,
                 const __bf16* __restrict__ A1, int w1,
                 const __bf16* __restrict__ A2, int w2,
                 const __bf16* __restrict__ Wih, int kih,
                 const __bf16* __restrict__ Whh,
                 const float* __restrict__ bih, const float* __restrict__ bhh,
                 float* __restrict__ c, float* __restrict__ hf,
                 __bf16* __restrict__ hb_out){
  __shared__ __align__(16) __bf16 Xs0[64*XS_STRIDE];
  __shared__ __align__(16) __bf16 Xs1[64*XS_STRIDE];
  __shared__ float gl[4][4][16][16];   // [gate][mtile][m][n]
  int tid  = threadIdx.x;
  int wv   = tid >> 5;
  int lane = tid & 31;
  int hi   = lane >> 4;
  int l15  = lane & 15;
  int mt   = wv & 3;
  int gh   = wv >> 2;
  int g0   = gh*2, g1 = gh*2 + 1;
  int arow = mt*16 + l15;

  v8f acc0 = {}, acc1 = {};
  const __bf16* w0ih = Wih + (size_t)(g0*HH + blockIdx.x*16 + l15)*kih;
  const __bf16* w1ih = Wih + (size_t)(g1*HH + blockIdx.x*16 + l15)*kih;
  gemm_region(A0, w0, w0ih,      w1ih,      Xs0, Xs1, tid, wv, hi, arow, acc0, acc1);
  gemm_region(A1, w1, w0ih + w0, w1ih + w0, Xs0, Xs1, tid, wv, hi, arow, acc0, acc1);
  const __bf16* w0hh = Whh + (size_t)(g0*HH + blockIdx.x*16 + l15)*w2;
  const __bf16* w1hh = Whh + (size_t)(g1*HH + blockIdx.x*16 + l15)*w2;
  gemm_region(A2, w2, w0hh, w1hh, Xs0, Xs1, tid, wv, hi, arow, acc0, acc1);

#pragma unroll
  for (int r=0; r<8; ++r){
    gl[g0][mt][r + hi*8][l15] = acc0[r];
    gl[g1][mt][r + hi*8][l15] = acc1[r];
  }
  __syncthreads();

  for (int idx=tid; idx<1024; idx+=256){
    int brow = idx >> 4, n = idx & 15;
    int mtl = brow >> 4, m = brow & 15;
    int hcol = blockIdx.x*16 + n;
    float ig = gl[0][mtl][m][n] + bih[hcol]        + bhh[hcol];
    float fg = gl[1][mtl][m][n] + bih[HH + hcol]   + bhh[HH + hcol];
    float gg = gl[2][mtl][m][n] + bih[2*HH + hcol] + bhh[2*HH + hcol];
    float og = gl[3][mtl][m][n] + bih[3*HH + hcol] + bhh[3*HH + hcol];
    float cold = c[brow*HH + hcol];
    float cn = sigf(fg)*cold + sigf(ig)*tanhf(gg);
    float hn = sigf(og)*tanhf(cn);
    c[brow*HH + hcol] = cn;
    if (hf) hf[brow*HH + hcol] = hn;
    hb_out[brow*HH + hcol] = (__bf16)hn;
  }
}

// Location-sensitive attention; one workgroup per batch row. block = 256.
__global__ __launch_bounds__(256)
void attn_kernel(const float* __restrict__ input, const unsigned char* __restrict__ mask,
                 const float* __restrict__ key, const float* __restrict__ h1f,
                 const float* __restrict__ Wq,  const float* __restrict__ lc,
                 const float* __restrict__ Wloc, const float* __restrict__ vatt,
                 float* __restrict__ aw, float* __restrict__ awc,
                 __bf16* __restrict__ ctxb, float* __restrict__ w_out, int t){
  int b = blockIdx.x;
  int tid = threadIdx.x;
  __shared__ float qs[MID_];
  __shared__ float aws[TIN_+30];
  __shared__ float awcs[TIN_+30];
  __shared__ float es[TIN_];
  __shared__ float red[TIN_];

  for (int i=tid; i<TIN_+30; i+=256){
    int ti = i - 15;
    bool inr = (ti >= 0) && (ti < TIN_);
    aws[i]  = inr ? aw[b*TIN_ + ti]  : 0.0f;
    awcs[i] = inr ? awc[b*TIN_ + ti] : 0.0f;
  }
  if (tid < MID_){
    const float* w = Wq + tid*HH;
    const float* h = h1f + b*HH;
    float s = 0.0f;
#pragma unroll 8
    for (int k=0;k<HH;++k) s += w[k]*h[k];
    qs[tid] = s;
  }
  __syncthreads();

  int ti = tid;
  float lf[32];
  for (int cf=0; cf<32; ++cf){
    const float* c0 = lc + cf*62;
    float s = 0.0f;
#pragma unroll
    for (int j=0;j<31;++j) s += c0[j]*aws[ti+j] + c0[31+j]*awcs[ti+j];
    lf[cf] = s;
  }
  const float* kb = key + ((size_t)b*TIN_ + ti)*MID_;
  float e = 0.0f;
  for (int m=0; m<MID_; ++m){
    const float* wl = Wloc + m*32;
    float lfm = 0.0f;
#pragma unroll
    for (int cch=0; cch<32; ++cch) lfm += wl[cch]*lf[cch];
    e += vatt[m]*tanhf(qs[m] + kb[m] + lfm);
  }
  if (!mask[b*TIN_ + ti]) e = -1e9f;
  es[ti] = e; red[ti] = e;
  __syncthreads();
  for (int s=128; s>0; s>>=1){ if (tid < s) red[tid] = fmaxf(red[tid], red[tid+s]); __syncthreads(); }
  float mx = red[0];
  __syncthreads();
  float ex = __expf(es[ti] - mx);
  red[ti] = ex;
  __syncthreads();
  for (int s=128; s>0; s>>=1){ if (tid < s) red[tid] += red[tid+s]; __syncthreads(); }
  float a2 = ex / red[0];
  __syncthreads();
  es[ti] = a2;
  aw[b*TIN_ + ti]  = a2;
  awc[b*TIN_ + ti] = awcs[15+ti] + a2;
  w_out[((size_t)b*TIN_ + ti)*TOUT_ + t] = a2;
  __syncthreads();
  const float* inb = input + (size_t)b*TIN_*BF_;
  for (int f=tid; f<BF_; f+=256){
    float s = 0.0f;
#pragma unroll 4
    for (int tj=0; tj<TIN_; ++tj) s += es[tj]*inb[(size_t)tj*BF_ + f];
    ctxb[b*BF_ + f] = (__bf16)s;
  }
}

// out[b, n, t] = [h2|ctx](1536) . Wout[n,:] + bout[n]
// grid = 5 (N tiles of 16 -> 80), block = 128 (wave = M-tile)
__global__ __launch_bounds__(128)
void out_kernel(const __bf16* __restrict__ h2b, const __bf16* __restrict__ ctxb,
                const __bf16* __restrict__ woutb, const float* __restrict__ bout,
                float* __restrict__ out, int t){
  int mt   = threadIdx.x >> 5;
  int lane = threadIdx.x & 31;
  int hi   = lane >> 4;
  int l15  = lane & 15;
  int ncol = blockIdx.x*16 + l15;
  int arow = mt*16 + l15;
  v8f acc = {};
  const __bf16* wrow = woutb + (size_t)ncol*1536;
  const __bf16* ha = h2b + (size_t)arow*HH;
  for (int k=0; k<1024; k+=32){
    acc = wmma_bf(frag16(ha + k, hi), frag16(wrow + k, hi), acc);
  }
  const __bf16* ca = ctxb + (size_t)arow*BF_;
  for (int k=0; k<512; k+=32){
    acc = wmma_bf(frag16(ca + k, hi), frag16(wrow + 1024 + k, hi), acc);
  }
#pragma unroll
  for (int r=0; r<8; ++r){
    int m = r + hi*8;
    int bg = mt*16 + m;
    int ng = blockIdx.x*16 + l15;
    out[((size_t)bg*NM_ + ng)*TOUT_ + t] = acc[r] + bout[ng];
  }
}

extern "C" void kernel_launch(void* const* d_in, const int* in_sizes, int n_in,
                              void* d_out, int out_size, void* d_ws, size_t ws_size,
                              hipStream_t stream) {
  const float* input   = (const float*)d_in[0];
  const unsigned char* mask = (const unsigned char*)d_in[1];
  const float* target  = (const float*)d_in[2];
  const float* W_pre1  = (const float*)d_in[3];
  const float* b_pre1  = (const float*)d_in[4];
  const float* W_pre2  = (const float*)d_in[5];
  const float* b_pre2  = (const float*)d_in[6];
  const float* Wq      = (const float*)d_in[7];
  const float* Wk      = (const float*)d_in[8];
  const float* lc      = (const float*)d_in[9];
  const float* Wloc    = (const float*)d_in[10];
  const float* vatt    = (const float*)d_in[11];
  const float* Wih1    = (const float*)d_in[12];
  const float* Whh1    = (const float*)d_in[13];
  const float* bih1    = (const float*)d_in[14];
  const float* bhh1    = (const float*)d_in[15];
  const float* Wih2    = (const float*)d_in[16];
  const float* Whh2    = (const float*)d_in[17];
  const float* bih2    = (const float*)d_in[18];
  const float* bhh2    = (const float*)d_in[19];
  const float* Wout    = (const float*)d_in[20];
  const float* bout    = (const float*)d_in[21];
  (void)in_sizes; (void)n_in; (void)out_size; (void)ws_size;

  char* ws = (char*)d_ws;
  size_t off = 0;
  auto carve = [&](size_t bytes)->char* {
    off = (off + 255) & ~(size_t)255;
    char* p = ws + off;
    off += bytes;
    return p;
  };

  __bf16* wih1b = (__bf16*)carve((size_t)4096*768*2);
  __bf16* whh1b = (__bf16*)carve((size_t)4096*1024*2);
  __bf16* wih2b = (__bf16*)carve((size_t)4096*1536*2);
  __bf16* whh2b = (__bf16*)carve((size_t)4096*1024*2);
  __bf16* woutb = (__bf16*)carve((size_t)80*1536*2);
  __bf16* xpre  = (__bf16*)carve((size_t)TOUT_*BSZ*PRE_*2);
  float*  key   = (float*)carve((size_t)BSZ*TIN_*MID_*4);
  // --- state block (zeroed each call) ---
  char* stateBegin = carve(0);
  float*  h1f  = (float*)carve((size_t)BSZ*HH*4);
  __bf16* h1b  = (__bf16*)carve((size_t)2*BSZ*HH*2);
  float*  c1   = (float*)carve((size_t)BSZ*HH*4);
  __bf16* h2b  = (__bf16*)carve((size_t)2*BSZ*HH*2);
  float*  c2   = (float*)carve((size_t)BSZ*HH*4);
  __bf16* ctxb = (__bf16*)carve((size_t)BSZ*BF_*2);
  float*  aw   = (float*)carve((size_t)BSZ*TIN_*4);
  float*  awc  = (float*)carve((size_t)BSZ*TIN_*4);
  char* stateEnd = carve(0);

  float* out_f = (float*)d_out;                       // [B, NM, TOUT]
  float* w_f   = out_f + (size_t)BSZ*NM_*TOUT_;       // [B, TIN, TOUT]

  {
    int n = (int)((stateEnd - stateBegin) / 4);
    zero_kernel<<<(n + 255)/256, 256, 0, stream>>>((float*)stateBegin, n);
  }
  {
    int n;
    n = 4096*768;  cvt_kernel<<<(n+255)/256,256,0,stream>>>(Wih1, wih1b, n);
    n = 4096*1024; cvt_kernel<<<(n+255)/256,256,0,stream>>>(Whh1, whh1b, n);
    n = 4096*1536; cvt_kernel<<<(n+255)/256,256,0,stream>>>(Wih2, wih2b, n);
    n = 4096*1024; cvt_kernel<<<(n+255)/256,256,0,stream>>>(Whh2, whh2b, n);
    n = 80*1536;   cvt_kernel<<<(n+255)/256,256,0,stream>>>(Wout, woutb, n);
  }
  prenet_kernel<<<TOUT_*BSZ, 256, 0, stream>>>(target, W_pre1, b_pre1, W_pre2, b_pre2, xpre);
  key_kernel<<<BSZ*TIN_, 128, 0, stream>>>(input, Wk, key);

  for (int t = 0; t < TOUT_; ++t){
    int cur = t & 1, prv = cur ^ 1;
    lstm_kernel<<<64, 256, 0, stream>>>(
        xpre + (size_t)t*BSZ*PRE_, PRE_,
        ctxb, BF_,
        h1b + (size_t)prv*BSZ*HH, HH,
        wih1b, PRE_ + BF_,
        whh1b, bih1, bhh1,
        c1, h1f, h1b + (size_t)cur*BSZ*HH);
    attn_kernel<<<BSZ, 256, 0, stream>>>(input, mask, key, h1f, Wq, lc, Wloc, vatt,
                                         aw, awc, ctxb, w_f, t);
    lstm_kernel<<<64, 256, 0, stream>>>(
        h1b + (size_t)cur*BSZ*HH, HH,
        ctxb, BF_,
        h2b + (size_t)prv*BSZ*HH, HH,
        wih2b, HH + BF_,
        whh2b, bih2, bhh2,
        c2, nullptr, h2b + (size_t)cur*BSZ*HH);
    out_kernel<<<5, 128, 0, stream>>>(h2b + (size_t)cur*BSZ*HH, ctxb, woutb, bout, out_f, t);
  }
}
